// DeformConvBlock_19756849561990
// MI455X (gfx1250) — compile-verified, matrix operand order checked by reading, not run on hardware
//
#include <hip/hip_runtime.h>
#include <hip/hip_bf16.h>
#include <math.h>

// ---------------- problem constants ----------------
constexpr int NB    = 4;
constexpr int NCIN  = 32;
constexpr int NCOUT = 64;
constexpr int NH    = 192;
constexpr int NW    = 192;
constexpr int NHW   = NH * NW;              // 36864
constexpr int NPIX  = NB * NHW;             // 147456
constexpr int NKS   = 9;                    // K-steps of 32 (K = 288)
constexpr int NNT   = 4;                    // N tiles of 16 (N = 64)
constexpr int NTILES = NPIX / 16;           // 9216 M-tiles
constexpr int NHO   = 96;
constexpr int NWO   = 96;
constexpr int NOUTE = NB * NCOUT * NHO * NWO; // 2359296
constexpr float FEPS = 1e-5f;

// ---------------- workspace layout (bytes) ----------------
constexpr size_t OFF_OFFS   = 0;
constexpr size_t SZ_OFFS    = (size_t)NB * 18 * NHW * 4;       // offsets f32
constexpr size_t OFF_Y      = OFF_OFFS + SZ_OFFS;
constexpr size_t SZ_Y       = (size_t)NB * NCOUT * NHW * 4;    // pre-BN y f32
constexpr size_t OFF_SUMBO  = OFF_Y + SZ_Y;                    // 256 f32
constexpr size_t OFF_SUMSQ  = OFF_SUMBO + 256 * 4;             // 64 f32
constexpr size_t OFF_AC     = OFF_SUMSQ + 64 * 4;              // 256 f32
constexpr size_t OFF_BC     = OFF_AC + 256 * 4;                // 256 f32
constexpr size_t OFF_BPACK  = OFF_BC + 256 * 4;                // bf16 B frags
constexpr size_t SZ_BPACK   = (size_t)NNT * NKS * 32 * 16 * 2; // 36 KB
constexpr size_t OFF_APANEL = OFF_BPACK + SZ_BPACK;            // bf16 A panels

// ---------------- WMMA types ----------------
typedef __attribute__((ext_vector_type(16))) __bf16 v16bf;
typedef __attribute__((ext_vector_type(8)))  float  v8f;

__device__ __forceinline__ __bf16 f2bf(float f) {
  unsigned u = __builtin_bit_cast(unsigned, f);
  unsigned r = u + 0x7fffu + ((u >> 16) & 1u);   // round-to-nearest-even
  unsigned short h = (unsigned short)(r >> 16);
  return __builtin_bit_cast(__bf16, h);
}

// ---------------- kernel Z: zero stat accumulators ----------------
__global__ void zero_stats_kernel(float* __restrict__ stats) {
  int i = blockIdx.x * 256 + threadIdx.x;
  if (i < 320) stats[i] = 0.f;   // sum_bo[256] + sumsq_o[64]
}

// ---------------- kernel A: offset conv (32 -> 18, 3x3 SAME) ----------------
__global__ void __launch_bounds__(256) offset_conv_kernel(
    const float* __restrict__ x, const float* __restrict__ w_off,
    const float* __restrict__ b_off, float* __restrict__ offs) {
  __shared__ float wsh[18 * NCIN * 9];
  __shared__ float bsh[18];
  int tid = threadIdx.x;
  for (int i = tid; i < 18 * NCIN * 9; i += 256) wsh[i] = w_off[i];
  if (tid < 18) bsh[tid] = b_off[tid];
  __syncthreads();
  int p = blockIdx.x * 256 + tid;
  if (p >= NPIX) return;
  int b = p / NHW;
  int hw = p - b * NHW;
  int h = hw / NW;
  int w = hw - h * NW;
  float acc[18];
#pragma unroll
  for (int oc = 0; oc < 18; ++oc) acc[oc] = bsh[oc];
  const float* xb = x + (size_t)b * NCIN * NHW;
  for (int ky = -1; ky <= 1; ++ky) {
    int yy = h + ky;
    if (yy < 0 || yy >= NH) continue;
    for (int kx = -1; kx <= 1; ++kx) {
      int xx = w + kx;
      if (xx < 0 || xx >= NW) continue;
      int tap = (ky + 1) * 3 + (kx + 1);
      float xv[NCIN];
#pragma unroll
      for (int c = 0; c < NCIN; ++c) xv[c] = xb[c * NHW + yy * NW + xx];
#pragma unroll 2
      for (int oc = 0; oc < 18; ++oc) {
        float a = acc[oc];
#pragma unroll
        for (int c = 0; c < NCIN; ++c)
          a = fmaf(xv[c], wsh[(oc * NCIN + c) * 9 + tap], a);
        acc[oc] = a;
      }
    }
  }
  for (int oc = 0; oc < 18; ++oc)
    offs[((size_t)b * 18 + oc) * NHW + hw] = acc[oc];
}

// ---------------- kernel W: pack w_def into WMMA B fragments (bf16) ----------
// B fragment lane layout (16-bit, 32x16): lane l holds column n = l&15;
// element i of lane l maps to k_local = ((i>>3)<<4) | ((l>>4)<<3) | (i&7).
__global__ void pack_weights_kernel(const float* __restrict__ w_def,
                                    __bf16* __restrict__ bpack) {
  int tid = blockIdx.x * blockDim.x + threadIdx.x;
  if (tid >= NNT * NKS * 32) return;
  int nt = tid / (NKS * 32);
  int s  = (tid / 32) % NKS;
  int l  = tid & 31;
  int o  = nt * 16 + (l & 15);
  int half = l >> 4;
#pragma unroll
  for (int i = 0; i < 16; ++i) {
    int kl = ((i >> 3) << 4) | (half << 3) | (i & 7);
    int f  = s * 32 + kl;             // feature = c*9 + kk  (c-major)
    int c  = f / 9;
    int kk = f - c * 9;
    float v = w_def[((size_t)o * NCIN + c) * 9 + kk];
    bpack[(((size_t)(nt * NKS + s) * 32 + l) << 4) + i] = f2bf(v);
  }
}

// ---------------- kernel S: bilinear sampling -> bf16 A panels ---------------
// One thread per (pixel, kernel-tap): computes bilinear weights once, then
// gathers all 32 channels and scatters them into the per-lane A-fragment
// layout (lane l holds row m = l&15, k subset selected by l>>4).
__global__ void __launch_bounds__(256) im2col_sample_kernel(
    const float* __restrict__ x, const float* __restrict__ offs,
    __bf16* __restrict__ apanel) {
  int idx = blockIdx.x * 256 + threadIdx.x;
  if (idx >= NPIX * 9) return;
  int p  = idx / 9;
  int kk = idx - p * 9;
  int b  = p / NHW;
  int hw = p - b * NHW;
  int h  = hw / NW;
  int w  = hw - h * NW;
  float dy = offs[((size_t)b * 18 + 2 * kk) * NHW + hw];
  float dx = offs[((size_t)b * 18 + 2 * kk + 1) * NHW + hw];
  float py = (float)(h + kk / 3 - 1) + dy;
  float px = (float)(w + kk % 3 - 1) + dx;
  float y0f = floorf(py), x0f = floorf(px);
  float wy = py - y0f, wx = px - x0f;
  int y0 = (int)y0f, x0 = (int)x0f;
  int y1 = y0 + 1, x1 = x0 + 1;
  float vy0 = (y0 >= 0 && y0 < NH) ? 1.f : 0.f;
  float vy1 = (y1 >= 0 && y1 < NH) ? 1.f : 0.f;
  float vx0 = (x0 >= 0 && x0 < NW) ? 1.f : 0.f;
  float vx1 = (x1 >= 0 && x1 < NW) ? 1.f : 0.f;
  int cy0 = min(max(y0, 0), NH - 1), cy1 = min(max(y1, 0), NH - 1);
  int cx0 = min(max(x0, 0), NW - 1), cx1 = min(max(x1, 0), NW - 1);
  float w00 = (1.f - wy) * (1.f - wx) * vy0 * vx0;
  float w01 = (1.f - wy) * wx * vy0 * vx1;
  float w10 = wy * (1.f - wx) * vy1 * vx0;
  float w11 = wy * wx * vy1 * vx1;
  int i00 = cy0 * NW + cx0, i01 = cy0 * NW + cx1;
  int i10 = cy1 * NW + cx0, i11 = cy1 * NW + cx1;
  int t = p >> 4, mloc = p & 15;
  const float* xb = x + (size_t)b * NCIN * NHW;
  for (int c = 0; c < NCIN; ++c) {
    const float* xc = xb + (size_t)c * NHW;
    float v = w00 * xc[i00] + w01 * xc[i01] + w10 * xc[i10] + w11 * xc[i11];
    int f = c * 9 + kk;
    int s = f >> 5;
    int kl = f & 31;
    int grp = kl >> 4, rem = kl & 15;
    int half = rem >> 3, within = rem & 7;
    int l = (half << 4) | mloc;
    int i = (grp << 3) | within;
    apanel[(((size_t)(t * NKS + s) * 32 + l) << 4) + i] = f2bf(v);
  }
}

// ---------------- kernel G: WMMA GEMM (M=NPIX, K=288, N=64) + stats ----------
__global__ void __launch_bounds__(256) wmma_gemm_kernel(
    const __bf16* __restrict__ apanel, const __bf16* __restrict__ bpack,
    const float* __restrict__ b_def, float* __restrict__ y,
    float* __restrict__ sum_bo, float* __restrict__ sumsq_o) {
  __shared__ __attribute__((aligned(32))) __bf16 Bsh[NNT * NKS * 32 * 16]; // 36 KB
  __shared__ float ls_sum[NCOUT], ls_sq[NCOUT];
  int tid = threadIdx.x;
  {  // stage all 36 B fragments into LDS (L2-resident, tiny)
    const uint4* src = (const uint4*)bpack;
    uint4* dst = (uint4*)Bsh;
#pragma unroll
    for (int k = 0; k < 9; ++k) dst[tid + k * 256] = src[tid + k * 256];
  }
  if (tid < NCOUT) { ls_sum[tid] = 0.f; ls_sq[tid] = 0.f; }
  __syncthreads();

  int wave = tid >> 5;
  int lane = tid & 31;
  int tile = blockIdx.x * 8 + wave;          // one 16-pixel M-tile per wave
  int bb   = (blockIdx.x * 128) / NHW;       // batch (128 | NHW)

  const __bf16* aptr = apanel + (((size_t)tile * NKS) * 32 + lane) * 16;

  v8f acc0 = {}, acc1 = {}, acc2 = {}, acc3 = {};
#pragma unroll
  for (int s = 0; s < NKS; ++s) {
    v16bf a = *(const v16bf*)(aptr + (size_t)s * 32 * 16);
    if (s + 1 < NKS)
      __builtin_prefetch(aptr + (size_t)(s + 1) * 32 * 16, 0, 1); // global_prefetch
    v16bf b0 = *(const v16bf*)&Bsh[((0 * NKS + s) * 32 + lane) * 16];
    v16bf b1 = *(const v16bf*)&Bsh[((1 * NKS + s) * 32 + lane) * 16];
    v16bf b2 = *(const v16bf*)&Bsh[((2 * NKS + s) * 32 + lane) * 16];
    v16bf b3 = *(const v16bf*)&Bsh[((3 * NKS + s) * 32 + lane) * 16];
    acc0 = __builtin_amdgcn_wmma_f32_16x16x32_bf16(false, a, false, b0, (short)0, acc0, false, false);
    acc1 = __builtin_amdgcn_wmma_f32_16x16x32_bf16(false, a, false, b1, (short)0, acc1, false, false);
    acc2 = __builtin_amdgcn_wmma_f32_16x16x32_bf16(false, a, false, b2, (short)0, acc2, false, false);
    acc3 = __builtin_amdgcn_wmma_f32_16x16x32_bf16(false, a, false, b3, (short)0, acc3, false, false);
  }

  // C/D layout: lane l holds column n = l&15; VGPR r holds row m = (l>=16?8:0)+r
  int mloc = (lane >> 4) << 3;
  size_t hw0 = (size_t)((tile * 16 + mloc) % NHW);
  int ncol = lane & 15;
#pragma unroll
  for (int nt = 0; nt < NNT; ++nt) {
    v8f acc = (nt == 0) ? acc0 : (nt == 1) ? acc1 : (nt == 2) ? acc2 : acc3;
    int o = nt * 16 + ncol;
    float bias = b_def[o];
    float* yp = y + ((size_t)(bb * NCOUT + o)) * NHW + hw0;
    float e0 = acc[0] + bias, e1 = acc[1] + bias, e2 = acc[2] + bias, e3 = acc[3] + bias;
    float e4 = acc[4] + bias, e5 = acc[5] + bias, e6 = acc[6] + bias, e7 = acc[7] + bias;
    *(float4*)(yp)     = make_float4(e0, e1, e2, e3);
    *(float4*)(yp + 4) = make_float4(e4, e5, e6, e7);
    float sm = e0 + e1 + e2 + e3 + e4 + e5 + e6 + e7;
    float sq = e0*e0 + e1*e1 + e2*e2 + e3*e3 + e4*e4 + e5*e5 + e6*e6 + e7*e7;
    atomicAdd(&ls_sum[o], sm);
    atomicAdd(&ls_sq[o], sq);
  }
  __syncthreads();
  if (tid < NCOUT) {
    atomicAdd(&sum_bo[bb * NCOUT + tid], ls_sum[tid]);
    atomicAdd(&sumsq_o[tid], ls_sq[tid]);
  }
}

// ---------------- kernel P: BN stats + SE MLP -> per-(b,o) affine ------------
__global__ void se_params_kernel(
    const float* __restrict__ sum_bo, const float* __restrict__ sumsq_o,
    const float* __restrict__ gamma, const float* __restrict__ beta,
    const float* __restrict__ fc1_w, const float* __restrict__ fc1_b,
    const float* __restrict__ fc2_w, const float* __restrict__ fc2_b,
    float* __restrict__ Acoef, float* __restrict__ Bcoef) {
  __shared__ float smean[NCOUT], srstd[NCOUT];
  __shared__ float ssv[NB][NCOUT];
  __shared__ float hsh[NB][4];
  int tid = threadIdx.x;           // 256 = NB * NCOUT
  int b = tid >> 6, o = tid & 63;
  if (tid < NCOUT) {
    float tot = 0.f;
    for (int bi = 0; bi < NB; ++bi) tot += sum_bo[bi * NCOUT + tid];
    float n = (float)(NB * NHW);
    float mean = tot / n;
    float var = sumsq_o[tid] / n - mean * mean;
    smean[tid] = mean;
    srstd[tid] = rsqrtf(var + FEPS);
  }
  __syncthreads();
  float mean = smean[o], rstd = srstd[o];
  float g = gamma[o], bt = beta[o];
  ssv[b][o] = g * rstd * (sum_bo[b * NCOUT + o] / (float)NHW - mean) + bt;
  __syncthreads();
  if (tid < NB * 4) {
    int bb = tid >> 2, j = tid & 3;
    float a = fc1_b[j];
    for (int oo = 0; oo < NCOUT; ++oo) a += ssv[bb][oo] * fc1_w[j * NCOUT + oo];
    hsh[bb][j] = a > 0.f ? a : 0.f;
  }
  __syncthreads();
  float z = fc2_b[o];
#pragma unroll
  for (int j = 0; j < 4; ++j) z += hsh[b][j] * fc2_w[o * 4 + j];
  float sig = 1.f / (1.f + __expf(-z));
  Acoef[b * NCOUT + o] = g * rstd * sig;
  Bcoef[b * NCOUT + o] = (bt - g * rstd * mean) * sig;
}

// ---------------- kernel F: affine + ReLU + 2x2 maxpool ----------------------
__global__ void pool_kernel(const float* __restrict__ y,
                            const float* __restrict__ Acoef,
                            const float* __restrict__ Bcoef,
                            float* __restrict__ out) {
  int idx = blockIdx.x * 256 + threadIdx.x;
  if (idx >= NOUTE) return;
  int wo = idx % NWO;
  int t  = idx / NWO;
  int ho = t % NHO;  t /= NHO;
  int o  = t % NCOUT;
  int b  = t / NCOUT;
  int co = b * NCOUT + o;
  float a = Acoef[co], bc = Bcoef[co];
  const float* yp = y + (size_t)co * NHW + (2 * ho) * NW + 2 * wo;
  float2 r0 = *(const float2*)(yp);
  float2 r1 = *(const float2*)(yp + NW);
  float v0 = fmaxf(r0.x * a + bc, 0.f);
  float v1 = fmaxf(r0.y * a + bc, 0.f);
  float v2 = fmaxf(r1.x * a + bc, 0.f);
  float v3 = fmaxf(r1.y * a + bc, 0.f);
  out[idx] = fmaxf(fmaxf(v0, v1), fmaxf(v2, v3));
}

// ---------------- host entry -------------------------------------------------
extern "C" void kernel_launch(void* const* d_in, const int* in_sizes, int n_in,
                              void* d_out, int out_size, void* d_ws, size_t ws_size,
                              hipStream_t stream) {
  (void)in_sizes; (void)n_in; (void)out_size; (void)ws_size;
  const float* x     = (const float*)d_in[0];
  const float* w_off = (const float*)d_in[1];
  const float* b_off = (const float*)d_in[2];
  const float* w_def = (const float*)d_in[3];
  const float* b_def = (const float*)d_in[4];
  const float* gamma = (const float*)d_in[5];
  const float* beta  = (const float*)d_in[6];
  const float* fc1_w = (const float*)d_in[7];
  const float* fc1_b = (const float*)d_in[8];
  const float* fc2_w = (const float*)d_in[9];
  const float* fc2_b = (const float*)d_in[10];
  float* out = (float*)d_out;

  char* ws = (char*)d_ws;
  float*  offs    = (float*)(ws + OFF_OFFS);
  float*  y       = (float*)(ws + OFF_Y);
  float*  sum_bo  = (float*)(ws + OFF_SUMBO);
  float*  sumsq_o = (float*)(ws + OFF_SUMSQ);
  float*  Acoef   = (float*)(ws + OFF_AC);
  float*  Bcoef   = (float*)(ws + OFF_BC);
  __bf16* bpack   = (__bf16*)(ws + OFF_BPACK);
  __bf16* apanel  = (__bf16*)(ws + OFF_APANEL);

  zero_stats_kernel<<<2, 256, 0, stream>>>(sum_bo);
  offset_conv_kernel<<<(NPIX + 255) / 256, 256, 0, stream>>>(x, w_off, b_off, offs);
  pack_weights_kernel<<<5, 256, 0, stream>>>(w_def, bpack);
  im2col_sample_kernel<<<(NPIX * 9 + 255) / 256, 256, 0, stream>>>(x, offs, apanel);
  wmma_gemm_kernel<<<NTILES / 8, 256, 0, stream>>>(apanel, bpack, b_def, y, sum_bo, sumsq_o);
  se_params_kernel<<<1, 256, 0, stream>>>(sum_bo, sumsq_o, gamma, beta,
                                          fc1_w, fc1_b, fc2_w, fc2_b, Acoef, Bcoef);
  pool_kernel<<<(NOUTE + 255) / 256, 256, 0, stream>>>(y, Acoef, Bcoef, out);
}
